// OA_Layer_90048284327975
// MI455X (gfx1250) — matrix operand order repaired and checked by reading, not compiled, not run on hardware
//
#include <hip/hip_runtime.h>
#include <hip/hip_bf16.h>

// ---------------------------------------------------------------------------
// OA (offset-attention) layer, B=8 C=256 N=4096 CQK=64, bf16 WMMA pipeline.
// All GEMM operands k-major (fragments = 1-2 vector loads); the dominant
// x_r GEMM uses 2x4 register blocking (8 WMMAs per 6 fragment loads).
// ---------------------------------------------------------------------------

#define B_ 8
#define C_ 256
#define N_ 4096
#define CQK_ 64
#define CN_ (C_ * N_)            // 1048576
#define BCN_ (B_ * C_ * N_)      // 8388608
#define NN_ ((size_t)N_ * N_)    // 16777216 per batch

typedef __attribute__((ext_vector_type(16))) __bf16 v16bf;
typedef __attribute__((ext_vector_type(8)))  __bf16 v8bf;
typedef __attribute__((ext_vector_type(8)))  float  v8f;

static __device__ __forceinline__ __bf16 f2bf(float f) { return (__bf16)f; }

static __device__ __forceinline__ v8f wmma_bf16(v16bf a, v16bf b, v8f c) {
  return __builtin_amdgcn_wmma_f32_16x16x32_bf16(
      /*neg_a=*/false, a, /*neg_b=*/false, b,
      /*c_mod=*/(short)0, c, /*reuse_a=*/false, /*reuse_b=*/false);
}

// A fragment, 16x32 bf16 from row-major [m][k] tile: two 16B vector loads.
static __device__ __forceinline__ v16bf load_a16x32(const __bf16* A, int lda, int lane) {
  int m  = lane & 15;
  int kb = (lane >> 4) * 16;
  const __bf16* p = A + m * lda + kb;
  v8bf lo = *(const v8bf*)(p);        // k = kb+0..7
  v8bf hi = *(const v8bf*)(p + 8);    // k = kb+8..15
  return __builtin_shufflevector(lo, hi, 0, 1, 2, 3, 4, 5, 6, 7,
                                 8, 9, 10, 11, 12, 13, 14, 15);
}

// B fragment, 32x16 bf16 from k-major storage B[k][n] = Bp[n*ldn + k]:
// one contiguous 32B vector load per lane.
static __device__ __forceinline__ v16bf load_b32x16(const __bf16* Bp, int ldn, int lane) {
  const __bf16* p = Bp + (lane & 15) * ldn + (lane >> 4) * 16;
  return *(const v16bf*)p;
}

// ---------------------------------------------------------------------------
// K0a: weights -> bf16
__global__ __launch_bounds__(256) void k_prep_w(const float* __restrict__ wqk,
                                                const float* __restrict__ wv,
                                                const float* __restrict__ wt,
                                                __bf16* __restrict__ wqkb,
                                                __bf16* __restrict__ wvb,
                                                __bf16* __restrict__ wtb) {
  int i = blockIdx.x * 256 + threadIdx.x;
  if (i < CQK_ * C_) {
    wqkb[i] = f2bf(wqk[i]);
  } else if (i < CQK_ * C_ + C_ * C_) {
    int j = i - CQK_ * C_;
    wvb[j] = f2bf(wv[j]);
  } else if (i < CQK_ * C_ + 2 * C_ * C_) {
    int j = i - CQK_ * C_ - C_ * C_;
    wtb[j] = f2bf(wt[j]);
  }
}

// K0b: xsT[b][n][c] = bf16(x + xyz), LDS-tiled 32x32 transpose.
__global__ __launch_bounds__(256) void k_prep_xs(const float* __restrict__ x,
                                                 const float* __restrict__ xyz,
                                                 __bf16* __restrict__ xsT) {
  __shared__ __bf16 tile[32][33];
  int b   = blockIdx.x >> 10;
  int rem = blockIdx.x & 1023;
  int c0  = (rem >> 7) << 5;      // 8 c-tiles
  int n0  = (rem & 127) << 5;     // 128 n-tiles
  int tr = threadIdx.x >> 5;      // 0..7
  int tn = threadIdx.x & 31;
#pragma unroll
  for (int i = 0; i < 4; ++i) {
    int c = tr + 8 * i;
    size_t idx = ((size_t)b * C_ + c0 + c) * N_ + n0 + tn;
    tile[c][tn] = f2bf(x[idx] + xyz[idx]);
  }
  __syncthreads();
#pragma unroll
  for (int i = 0; i < 4; ++i) {
    int nr = tr + 8 * i;
    xsT[((size_t)b * N_ + n0 + nr) * C_ + c0 + tn] = tile[tn][nr];
  }
}

// ---------------------------------------------------------------------------
// K1: fused Q / V projection. mt<4 -> Q rows (wqk), else V rows (wv).
// Q written transposed [b][n][64]; V row-major [b][c][n] (A of x_r GEMM).
__global__ __launch_bounds__(256) void k_proj(const __bf16* __restrict__ wqkb,
                                              const __bf16* __restrict__ wvb,
                                              const __bf16* __restrict__ xsT,
                                              const float* __restrict__ bqk,
                                              const float* __restrict__ bv,
                                              __bf16* __restrict__ qT,
                                              __bf16* __restrict__ vmat) {
  int wave = threadIdx.x >> 5, lane = threadIdx.x & 31;
  int gid = blockIdx.x * 8 + wave;            // 40960 tile-waves
  int b   = gid / 5120;
  int rem = gid % 5120;
  int mt  = rem >> 8;
  int n0  = (rem & 255) << 4;
  bool isq = (mt < 4);
  const __bf16* A = isq ? wqkb : wvb;
  int m0 = isq ? (mt << 4) : ((mt - 4) << 4);
  const __bf16* Bb = xsT + (size_t)b * CN_;   // k-major: [n][c]

  v8f acc = {};
#pragma unroll
  for (int kt = 0; kt < 8; ++kt) {
    int k0 = kt << 5;
    v16bf af = load_a16x32(A + m0 * C_ + k0, C_, lane);
    v16bf bf = load_b32x16(Bb + (size_t)n0 * C_ + k0, C_, lane);
    acc = wmma_bf16(af, bf, acc);
  }
  int hi = lane >> 4, ln = lane & 15, nn = n0 + ln;
  if (isq) {
    v8bf st;
#pragma unroll
    for (int r = 0; r < 8; ++r) st[r] = f2bf(acc[r] + bqk[m0 + 8 * hi + r]);
    *(v8bf*)(qT + ((size_t)b * N_ + nn) * CQK_ + m0 + 8 * hi) = st;  // 16B store
  } else {
#pragma unroll
    for (int r = 0; r < 8; ++r) {
      int row = m0 + 8 * hi + r;
      vmat[((size_t)b * C_ + row) * N_ + nn] = f2bf(acc[r] + bv[row]);
    }
  }
}

// ---------------------------------------------------------------------------
// K2: energy = QtQ row-block + online row softmax; writes exp(e-M)/L bf16
// TRANSPOSED: attnT[b][j][i] (16B vector store per lane; k-major for x_r).
__global__ __launch_bounds__(256) void k_energy(const __bf16* __restrict__ qT,
                                                __bf16* __restrict__ attnT) {
  int wave = threadIdx.x >> 5, lane = threadIdx.x & 31;
  int hi = lane >> 4, ln = lane & 15;
  int b  = blockIdx.x >> 8;
  int i0 = (blockIdx.x & 255) << 4;
  const __bf16* qb = qT + (size_t)b * N_ * CQK_;
  __bf16* attnb = attnT + (size_t)b * NN_;

  v16bf a0 = load_a16x32(qb + (size_t)i0 * CQK_ + 0,  CQK_, lane);
  v16bf a1 = load_a16x32(qb + (size_t)i0 * CQK_ + 32, CQK_, lane);

  float mrow[8], lrow[8];
#pragma unroll
  for (int r = 0; r < 8; ++r) { mrow[r] = -3.4e38f; lrow[r] = 0.0f; }

  // Pass 1: online max / sum-of-exp over this wave's 32 column tiles.
  for (int t = 0; t < 32; ++t) {
    int j0 = (wave * 32 + t) << 4;
    v16bf b0 = load_b32x16(qb + (size_t)j0 * CQK_ + 0,  CQK_, lane);
    v16bf b1 = load_b32x16(qb + (size_t)j0 * CQK_ + 32, CQK_, lane);
    v8f acc = {};
    acc = wmma_bf16(a0, b0, acc);
    acc = wmma_bf16(a1, b1, acc);
#pragma unroll
    for (int r = 0; r < 8; ++r) {
      float e = acc[r];
      float tm = e;
#pragma unroll
      for (int s = 1; s <= 8; s <<= 1) tm = fmaxf(tm, __shfl_xor(tm, s, 32));
      float mnew = fmaxf(mrow[r], tm);
      float p = __expf(e - mnew);
      float ps = p;
#pragma unroll
      for (int s = 1; s <= 8; s <<= 1) ps += __shfl_xor(ps, s, 32);
      lrow[r] = lrow[r] * __expf(mrow[r] - mnew) + ps;
      mrow[r] = mnew;
    }
  }

  // Merge per-wave stats across the 8 waves.
  __shared__ float sm[8][16];
  __shared__ float sl[8][16];
  if (ln == 0) {
#pragma unroll
    for (int r = 0; r < 8; ++r) {
      sm[wave][hi * 8 + r] = mrow[r];
      sl[wave][hi * 8 + r] = lrow[r];
    }
  }
  __syncthreads();
  float Mrow[8], Linv[8];
#pragma unroll
  for (int r = 0; r < 8; ++r) {
    int row = hi * 8 + r;
    float M = -3.4e38f;
#pragma unroll
    for (int w = 0; w < 8; ++w) M = fmaxf(M, sm[w][row]);
    float L = 0.0f;
#pragma unroll
    for (int w = 0; w < 8; ++w) L += sl[w][row] * __expf(sm[w][row] - M);
    Mrow[r] = M;
    Linv[r] = 1.0f / L;
  }

  // Pass 2: recompute tiles; transposed 16B vector store per lane.
  for (int t = 0; t < 32; ++t) {
    int j0 = (wave * 32 + t) << 4;
    v16bf b0 = load_b32x16(qb + (size_t)j0 * CQK_ + 0,  CQK_, lane);
    v16bf b1 = load_b32x16(qb + (size_t)j0 * CQK_ + 32, CQK_, lane);
    v8f acc = {};
    acc = wmma_bf16(a0, b0, acc);
    acc = wmma_bf16(a1, b1, acc);
    v8bf st;
#pragma unroll
    for (int r = 0; r < 8; ++r)
      st[r] = f2bf(__expf(acc[r] - Mrow[r]) * Linv[r]);
    // attnT[j0+ln][i0 + 8*hi + 0..7]
    *(v8bf*)(attnb + (size_t)(j0 + ln) * N_ + i0 + 8 * hi) = st;
  }
}

// ---------------------------------------------------------------------------
// K3: reciprocal of column sums (rows of attnT): wave-per-column reduction.
__global__ __launch_bounds__(256) void k_colsum(const __bf16* __restrict__ attnT,
                                                float* __restrict__ recip) {
  int wave = threadIdx.x >> 5, lane = threadIdx.x & 31;
  int g = blockIdx.x * 8 + wave;            // 32768 = B * N columns
  int b = g >> 12, j = g & (N_ - 1);
  const __bf16* row = attnT + (size_t)b * NN_ + (size_t)j * N_;
  float s = 0.0f;
#pragma unroll
  for (int it = 0; it < 16; ++it) {
    v8bf v = *(const v8bf*)(row + (it * 32 + lane) * 8);   // coalesced 16B
#pragma unroll
    for (int e = 0; e < 8; ++e) s += (float)v[e];
  }
#pragma unroll
  for (int m = 1; m <= 16; m <<= 1) s += __shfl_xor(s, m, 32);
  if (lane == 0) recip[g] = 1.0f / (1e-9f + s);
}

// ---------------------------------------------------------------------------
// K4: x_r = V * attn (K = 4096), column renorm fused; writes
// diffT[b][n][c] = xs - x_r. 2x4 register blocking: each wave owns a
// 32x64 output block -> 8 WMMAs per 6 fragment loads per k-step.
__global__ __launch_bounds__(256) void k_xr(const __bf16* __restrict__ vmat,
                                            const __bf16* __restrict__ attnT,
                                            const float* __restrict__ recip,
                                            const float* __restrict__ x,
                                            const float* __restrict__ xyz,
                                            __bf16* __restrict__ diffT) {
  int wave = threadIdx.x >> 5, lane = threadIdx.x & 31;
  int gid = blockIdx.x * 8 + wave;          // 4096 block-waves
  int b   = gid >> 9;
  int rem = gid & 511;
  int c0  = (rem >> 6) << 5;                // 8 m-blocks of 32 rows
  int n0  = (rem & 63) << 6;                // 64 n-blocks of 64 cols
  const __bf16* Ab = vmat  + (size_t)b * CN_;  // [c][n] row-major (K along n)
  const __bf16* Bb = attnT + (size_t)b * NN_;  // k-major: [m][n(k)]

  v8f acc[2][4] = {};
  for (int kt = 0; kt < 128; ++kt) {
    int k0 = kt << 5;
    if (kt + 1 < 128) {  // prefetch next k-tiles into cache
      __builtin_prefetch(Ab + (size_t)c0 * N_ + k0 + 32, 0, 3);
      __builtin_prefetch(Bb + (size_t)n0 * N_ + k0 + 32, 0, 3);
    }
    v16bf a0 = load_a16x32(Ab + (size_t)c0 * N_ + k0, N_, lane);
    v16bf a1 = load_a16x32(Ab + (size_t)(c0 + 16) * N_ + k0, N_, lane);
    v16bf bf[4];
#pragma unroll
    for (int jt = 0; jt < 4; ++jt)
      bf[jt] = load_b32x16(Bb + (size_t)(n0 + 16 * jt) * N_ + k0, N_, lane);
#pragma unroll
    for (int jt = 0; jt < 4; ++jt) {
      acc[0][jt] = wmma_bf16(a0, bf[jt], acc[0][jt]);
      acc[1][jt] = wmma_bf16(a1, bf[jt], acc[1][jt]);
    }
  }
  int hi = lane >> 4, ln = lane & 15;
#pragma unroll
  for (int jt = 0; jt < 4; ++jt) {
    int nn = n0 + 16 * jt + ln;
    float rc = recip[b * N_ + nn];
#pragma unroll
    for (int it = 0; it < 2; ++it) {
      v8bf st;
#pragma unroll
      for (int r = 0; r < 8; ++r) {
        int cc = c0 + 16 * it + 8 * hi + r;
        size_t idx = ((size_t)b * C_ + cc) * N_ + nn;
        st[r] = f2bf(x[idx] + xyz[idx] - acc[it][jt][r] * rc);
      }
      // diffT[nn][c0 + 16*it + 8*hi + 0..7]
      *(v8bf*)(diffT + ((size_t)b * N_ + nn) * C_ + c0 + 16 * it + 8 * hi) = st;
    }
  }
}

// ---------------------------------------------------------------------------
// K5: t = wt * diff + bt; fused BatchNorm + LeakyReLU(0.2) + residual.
// 1x4 register blocking (16x64 output per wave).
__global__ __launch_bounds__(256) void k_final(const __bf16* __restrict__ wtb,
                                               const __bf16* __restrict__ diffT,
                                               const float* __restrict__ bt,
                                               const float* __restrict__ gamma,
                                               const float* __restrict__ beta,
                                               const float* __restrict__ mean,
                                               const float* __restrict__ var,
                                               const float* __restrict__ x,
                                               const float* __restrict__ xyz,
                                               float* __restrict__ out) {
  int wave = threadIdx.x >> 5, lane = threadIdx.x & 31;
  int gid = blockIdx.x * 8 + wave;          // 8192 block-waves
  int b   = gid >> 10;
  int rem = gid & 1023;
  int o0  = (rem >> 6) << 4;                // 16 o-tiles
  int n0  = (rem & 63) << 6;                // 64 n-blocks of 64 cols
  const __bf16* Bb = diffT + (size_t)b * CN_;  // k-major: [n][c(k)]

  v8f acc[4] = {};
#pragma unroll
  for (int kt = 0; kt < 8; ++kt) {
    int k0 = kt << 5;
    v16bf af = load_a16x32(wtb + o0 * C_ + k0, C_, lane);
#pragma unroll
    for (int jt = 0; jt < 4; ++jt) {
      v16bf bf = load_b32x16(Bb + (size_t)(n0 + 16 * jt) * C_ + k0, C_, lane);
      acc[jt] = wmma_bf16(af, bf, acc[jt]);
    }
  }
  int hi = lane >> 4, ln = lane & 15;
#pragma unroll
  for (int jt = 0; jt < 4; ++jt) {
    int nn = n0 + 16 * jt + ln;
#pragma unroll
    for (int r = 0; r < 8; ++r) {
      int o = o0 + 8 * hi + r;
      size_t idx = ((size_t)b * C_ + o) * N_ + nn;
      float t  = acc[jt][r] + bt[o];
      float sc = rsqrtf(var[o] + 1e-5f) * gamma[o];
      t = (t - mean[o]) * sc + beta[o];
      t = (t >= 0.0f) ? t : 0.2f * t;       // LeakyReLU(0.2)
      out[idx] = x[idx] + xyz[idx] + t;
    }
  }
}

// ---------------------------------------------------------------------------
extern "C" void kernel_launch(void* const* d_in, const int* in_sizes, int n_in,
                              void* d_out, int out_size, void* d_ws, size_t ws_size,
                              hipStream_t stream) {
  (void)in_sizes; (void)n_in; (void)out_size; (void)ws_size;
  const float* x     = (const float*)d_in[0];
  const float* xyz   = (const float*)d_in[1];
  const float* wqk   = (const float*)d_in[2];
  const float* bqk   = (const float*)d_in[3];
  const float* wv    = (const float*)d_in[4];
  const float* bv    = (const float*)d_in[5];
  const float* wt    = (const float*)d_in[6];
  const float* bt    = (const float*)d_in[7];
  const float* gamma = (const float*)d_in[8];
  const float* beta  = (const float*)d_in[9];
  const float* mean  = (const float*)d_in[10];
  const float* var   = (const float*)d_in[11];
  float* out = (float*)d_out;

  // Workspace layout (bytes, 256-aligned):
  char* w = (char*)d_ws;
  __bf16* wqkb  = (__bf16*)(w + 0);                    //    32768 B
  __bf16* wvb   = (__bf16*)(w + 32768);                //   131072 B
  __bf16* wtb   = (__bf16*)(w + 163840);               //   131072 B
  __bf16* xsT   = (__bf16*)(w + 294912);               // 16777216 B  [b][n][c]
  __bf16* qT    = (__bf16*)(w + 17072128);             //  4194304 B  [b][n][64]
  __bf16* vmat  = (__bf16*)(w + 21266432);             // 16777216 B  [b][c][n]
  __bf16* attnT = (__bf16*)(w + 38043648);             // 268435456 B [b][j][i]
  float*  recp  = (float*) (w + 306479104);            //   131072 B  [b][j]
  __bf16* diffT = (__bf16*)(w + 306610176);            // 16777216 B  [b][n][c]

  k_prep_w <<<576,   256, 0, stream>>>(wqk, wv, wt, wqkb, wvb, wtb);
  k_prep_xs<<<8192,  256, 0, stream>>>(x, xyz, xsT);
  k_proj   <<<5120,  256, 0, stream>>>(wqkb, wvb, xsT, bqk, bv, qT, vmat);
  k_energy <<<2048,  256, 0, stream>>>(qT, attnT);
  k_colsum <<<4096,  256, 0, stream>>>(attnT, recp);
  k_xr     <<<512,   256, 0, stream>>>(vmat, attnT, recp, x, xyz, diffT);
  k_final  <<<1024,  256, 0, stream>>>(wtb, diffT, bt, gamma, beta, mean, var, x, xyz, out);
}
